// PLE_38036230374040
// MI455X (gfx1250) — matrix operand order repaired and checked by reading, not compiled, not run on hardware
//
#include <hip/hip_runtime.h>
#include <hip/hip_bf16.h>

// ---------------- types for WMMA ----------------
typedef __attribute__((ext_vector_type(16))) __bf16 v16bf;
typedef __attribute__((ext_vector_type(8)))  float  v8f;
typedef __attribute__((ext_vector_type(4)))  unsigned int uint4v;

union Frag {
    uint4v u[2];
    v16bf  v;
};

// Problem constants
#define HH 768
#define EE 8
#define LL 2
#define ROWS 4096            // B*S
#define NN   6144            // H*E
#define KT   24              // 768/32 k-tiles
#define NT   384             // 6144/16 n-tiles

__device__ __forceinline__ unsigned short f2bf(float f) {
    unsigned int u = __float_as_uint(f);
    unsigned int r = u + 0x7FFFu + ((u >> 16) & 1u);   // round-to-nearest-even
    return (unsigned short)(r >> 16);
}

// CDNA5 async global->LDS copy (16B per lane), tracked by ASYNCcnt
__device__ __forceinline__ void async_g2lds_b128(unsigned int lds_off, const void* gaddr) {
    asm volatile("global_load_async_to_lds_b128 %0, %1, off"
                 :: "v"(lds_off), "v"(gaddr) : "memory");
}
__device__ __forceinline__ void wait_asynccnt0() {
    asm volatile("s_wait_asynccnt 0" ::: "memory");
}

// ---------------- weight convert + swizzle ----------------
// out flat layout: [lt(0..5)][nt(0..383)][kt(0..23)][lane(0..31)][j(0..7)]  (uint = packed bf16 pair)
// lane: n = nt*16 + (lane&15), k0 = kt*32 + (lane>>4)*16 + 2j  -> B-matrix fragment layout
__global__ void ple_convert_w(const float* __restrict__ W, unsigned int* __restrict__ out, int total) {
    int idx = blockIdx.x * blockDim.x + threadIdx.x;
    if (idx >= total) return;
    int j    = idx & 7;
    int lane = (idx >> 3) & 31;
    int kt   = (idx >> 8) % KT;
    int rest = (idx >> 8) / KT;
    int nt   = rest % NT;
    int lt   = rest / NT;                       // l*3 + t
    int n  = nt * 16 + (lane & 15);
    int k0 = kt * 32 + (lane >> 4) * 16 + j * 2;
    size_t base = (size_t)lt * HH * NN;
    float f0 = W[base + (size_t)k0 * NN + n];
    float f1 = W[base + (size_t)(k0 + 1) * NN + n];
    out[idx] = (unsigned int)f2bf(f0) | ((unsigned int)f2bf(f1) << 16);
}

// ---------------- init residual buffers ----------------
__global__ void ple_init_h(const float* __restrict__ a, const float* __restrict__ b,
                           const float* __restrict__ c, float* __restrict__ h, int n) {
    int i = blockIdx.x * blockDim.x + threadIdx.x;
    if (i < n) { h[i] = a[i]; h[n + i] = b[i]; h[2 * n + i] = c[i]; }
}

// ---------------- layernorm (wave per row) ----------------
__global__ void __launch_bounds__(256)
ple_layernorm(const float* __restrict__ h, const float* __restrict__ gam,
              const float* __restrict__ bet, float* __restrict__ xf,
              unsigned short* __restrict__ xb) {
    int wave = threadIdx.x >> 5, lane = threadIdx.x & 31;
    int row = blockIdx.x * 8 + wave;        // 0..12287 (3 towers * 4096)
    int tower = row >> 12;
    const float* hp = h + (size_t)row * HH;
    float vals[24];
    float s = 0.f, sq = 0.f;
#pragma unroll
    for (int i = 0; i < 24; ++i) {
        float v = hp[lane + 32 * i];
        vals[i] = v; s += v; sq += v * v;
    }
#pragma unroll
    for (int o = 16; o > 0; o >>= 1) { s += __shfl_xor(s, o, 32); sq += __shfl_xor(sq, o, 32); }
    float mu  = s * (1.f / HH);
    float var = sq * (1.f / HH) - mu * mu;
    float rs  = rsqrtf(var + 1e-5f);
    const float* gp = gam + tower * HH;
    const float* bp = bet + tower * HH;
    float* xfp = xf + (size_t)row * HH;
    unsigned short* xbp = xb + (size_t)row * HH;
#pragma unroll
    for (int i = 0; i < 24; ++i) {
        int k = lane + 32 * i;
        float v = gp[k] * (vals[i] - mu) * rs + bp[k];
        xfp[k] = v;
        xbp[k] = f2bf(v);
    }
}

// ---------------- gates: dot + softmax (wave per row) ----------------
__global__ void __launch_bounds__(256)
ple_gates(const float* __restrict__ xf,
          const float* __restrict__ Wsh, const float* __restrict__ bsh,
          const float* __restrict__ Wt1, const float* __restrict__ bt1,
          const float* __restrict__ Wt2, const float* __restrict__ bt2,
          float* __restrict__ g) {
    int wave = threadIdx.x >> 5, lane = threadIdx.x & 31;
    int row  = blockIdx.x * 8 + wave;        // 0..4095
    int type = blockIdx.y;                   // 0=sh(24) 1=t1(16) 2=t2(16)
    int width = (type == 0) ? 24 : 16;
    const float* W  = (type == 0) ? Wsh : ((type == 1) ? Wt1 : Wt2);
    const float* bb = (type == 0) ? bsh : ((type == 1) ? bt1 : bt2);
    const float* xp = xf + ((size_t)type * ROWS + row) * HH;
    float acc = (lane < width) ? bb[lane] : -__builtin_inff();
    if (lane < width) {
        for (int k = 0; k < HH; ++k) acc = fmaf(xp[k], W[k * width + lane], acc);
    }
    float m = acc;
#pragma unroll
    for (int o = 16; o > 0; o >>= 1) m = fmaxf(m, __shfl_xor(m, o, 32));
    float ev = (lane < width) ? __expf(acc - m) : 0.f;
    float ssum = ev;
#pragma unroll
    for (int o = 16; o > 0; o >>= 1) ssum += __shfl_xor(ssum, o, 32);
    if (lane < width) {
        int off = (type == 0) ? 0 : ((type == 1) ? 24 : 40);
        g[(size_t)row * 56 + off + lane] = ev / ssum;
    }
}

// ---------------- WMMA expert GEMM + bias + exact GELU ----------------
// Block tile: M=32, N=512. 8 waves; each wave: 2 M-tiles x 4 N-tiles (8 WMMA / k-step).
// A tile staged in LDS (swizzled to A-fragment layout) via async global->LDS b128.
// B fragments double-buffered from the pre-swizzled global weight buffer.
__global__ void __launch_bounds__(256)
ple_gemm_experts(const unsigned short* __restrict__ xb,   // [3][4096][768] bf16
                 const unsigned int* __restrict__ wsw,    // layer slice of swizzled weights
                 const float* __restrict__ bexp,          // [3][6144] layer slice
                 float* __restrict__ e) {                 // [3][4096][6144]
    __shared__ unsigned int aswz[2 * KT * 32 * 8];        // 48KB: [mt][kt][lane][j]

    const int tid  = threadIdx.x;
    const int lane = tid & 31;
    const int wave = tid >> 5;

    const int mblk  = blockIdx.x;           // 0..383 (12288 rows / 32)
    const int tower = mblk >> 7;            // 128 m-blocks per tower
    const int mbase = (mblk & 127) * 32;
    const int ntWave = blockIdx.y * 32 + wave * 4;   // n-tile base for this wave

    // ---- async fill of swizzled A tile: 3072 x 16B chunks, 12 per thread ----
    const unsigned short* xtow = xb + ((size_t)tower * ROWS + mbase) * HH;
#pragma unroll
    for (int it = 0; it < 12; ++it) {
        int c   = tid + it * 256;           // 0..3071
        int row = c / 96;                   // 0..31
        int kk  = c % 96;                   // 8-bf16 (16B) chunk within row
        int mt   = row >> 4;
        int kt   = kk >> 2;
        int lsub = (((kk >> 1) & 1) << 4) + (row & 15);   // khalf*16 + m
        int j    = (kk & 1) ? 4 : 0;               // k-group
        unsigned int ldsoff =
            (unsigned int)(uintptr_t)&aswz[(((mt * KT + kt) * 32 + lsub) << 3) + j];
        async_g2lds_b128(ldsoff, (const void*)(xtow + (size_t)row * HH + kk * 8));
    }
    wait_asynccnt0();
    __syncthreads();

    // ---- main K loop: A from LDS, B double-buffered from global ----
    const unsigned int* wb0 = wsw + ((size_t)(tower * NT + ntWave) * KT * 32) * 8 + lane * 8;
    // B frag for (t4, kt): wb0 + (t4*KT + kt)*32*8

    v8f acc[2][4] = {};
    Frag bbuf[2][4];
#pragma unroll
    for (int t4 = 0; t4 < 4; ++t4) {
        const unsigned int* bp = wb0 + ((size_t)t4 * KT) * 256;
        bbuf[0][t4].u[0] = *(const uint4v*)(bp);
        bbuf[0][t4].u[1] = *(const uint4v*)(bp + 4);
    }
#pragma unroll 2
    for (int kt = 0; kt < KT; ++kt) {
        const int cur = kt & 1, nxt = cur ^ 1;
        if (kt < KT - 1) {
#pragma unroll
            for (int t4 = 0; t4 < 4; ++t4) {
                const unsigned int* bp = wb0 + ((size_t)t4 * KT + kt + 1) * 256;
                bbuf[nxt][t4].u[0] = *(const uint4v*)(bp);
                bbuf[nxt][t4].u[1] = *(const uint4v*)(bp + 4);
            }
        }
        Frag a0, a1;
        const unsigned int* l0 = &aswz[((0 * KT + kt) * 32 + lane) << 3];
        const unsigned int* l1 = &aswz[((1 * KT + kt) * 32 + lane) << 3];
        a0.u[0] = *(const uint4v*)(l0); a0.u[1] = *(const uint4v*)(l0 + 4);
        a1.u[0] = *(const uint4v*)(l1); a1.u[1] = *(const uint4v*)(l1 + 4);
#pragma unroll
        for (int t4 = 0; t4 < 4; ++t4) {
            acc[0][t4] = __builtin_amdgcn_wmma_f32_16x16x32_bf16(
                false, a0.v, false, bbuf[cur][t4].v, (short)0, acc[0][t4], false, false);
            acc[1][t4] = __builtin_amdgcn_wmma_f32_16x16x32_bf16(
                false, a1.v, false, bbuf[cur][t4].v, (short)0, acc[1][t4], false, false);
        }
    }

    // ---- epilogue: bias + exact GELU, C layout: VGPR r -> M = r + 8*khalf, N = lane&15 ----
    const int khalf = lane >> 4;
    const int nCol  = lane & 15;
    float* ebase = e + ((size_t)tower * ROWS + mbase) * NN;
#pragma unroll
    for (int mt = 0; mt < 2; ++mt) {
#pragma unroll
        for (int t4 = 0; t4 < 4; ++t4) {
            int nFull = (ntWave + t4) * 16 + nCol;
            float bias = bexp[tower * NN + nFull];
#pragma unroll
            for (int r = 0; r < 8; ++r) {
                float v = acc[mt][t4][r] + bias;
                v = 0.5f * v * (1.0f + erff(v * 0.70710678118654752f));   // exact gelu
                ebase[(size_t)(mt * 16 + khalf * 8 + r) * NN + nFull] = v;
            }
        }
    }
}

// ---------------- gated combine + residual (block per row) ----------------
__global__ void __launch_bounds__(256)
ple_combine(const float* __restrict__ e, const float* __restrict__ g,
            float* __restrict__ h) {
    int row = blockIdx.x;
    __shared__ float gs[56];
    if (threadIdx.x < 56) gs[threadIdx.x] = g[(size_t)row * 56 + threadIdx.x];
    __syncthreads();
    const float* e_sh = e + ((size_t)0 * ROWS + row) * NN;
    const float* e_t1 = e + ((size_t)1 * ROWS + row) * NN;
    const float* e_t2 = e + ((size_t)2 * ROWS + row) * NN;
    for (int x = threadIdx.x; x < HH; x += 256) {
        const float* p1 = e_t1 + x * EE;
        const float* p0 = e_sh + x * EE;
        const float* p2 = e_t2 + x * EE;
        float o_sh = 0.f, o_t1 = 0.f, o_t2 = 0.f;
#pragma unroll
        for (int j = 0; j < EE; ++j) {
            float v1 = p1[j], v0 = p0[j], v2 = p2[j];
            o_sh = fmaf(gs[j],      v1, o_sh);
            o_sh = fmaf(gs[8 + j],  v0, o_sh);
            o_sh = fmaf(gs[16 + j], v2, o_sh);
            o_t1 = fmaf(gs[24 + j], v1, o_t1);
            o_t1 = fmaf(gs[32 + j], v0, o_t1);
            o_t2 = fmaf(gs[40 + j], v2, o_t2);
            o_t2 = fmaf(gs[48 + j], v0, o_t2);
        }
        h[((size_t)0 * ROWS + row) * HH + x] += o_sh;
        h[((size_t)1 * ROWS + row) * HH + x] += o_t1;
        h[((size_t)2 * ROWS + row) * HH + x] += o_t2;
    }
}

// ---------------- output heads ----------------
__global__ void ple_heads(const float* __restrict__ h,
                          const float* __restrict__ Wsp, const float* __restrict__ bsp,
                          const float* __restrict__ Was, const float* __restrict__ bas,
                          const float* __restrict__ Wae, const float* __restrict__ bae,
                          float* __restrict__ out) {
    int row = blockIdx.x;
    int t = threadIdx.x;
    if (t >= 67) return;
    const float* ht1 = h + ((size_t)1 * ROWS + row) * HH;
    const float* ht2 = h + ((size_t)2 * ROWS + row) * HH;
    if (t < 5) {
        float a = bsp[t];
        for (int k = 0; k < HH; ++k) a = fmaf(ht1[k], Wsp[k * 5 + t], a);
        out[(size_t)row * 5 + t] = a;
    } else if (t < 36) {
        int j = t - 5;
        float a = bas[j];
        for (int k = 0; k < HH; ++k) a = fmaf(ht2[k], Was[k * 31 + j], a);
        out[20480 + (size_t)row * 31 + j] = a;
    } else {
        int j = t - 36;
        float a = bae[j];
        for (int k = 0; k < HH; ++k) a = fmaf(ht2[k], Wae[k * 31 + j], a);
        out[147456 + (size_t)row * 31 + j] = a;
    }
}

extern "C" void kernel_launch(void* const* d_in, const int* in_sizes, int n_in,
                              void* d_out, int out_size, void* d_ws, size_t ws_size,
                              hipStream_t stream) {
    const float* share_hidden = (const float*)d_in[0];
    const float* span_hidden  = (const float*)d_in[1];
    const float* attr_hidden  = (const float*)d_in[2];
    const float* ln_g  = (const float*)d_in[3];
    const float* ln_b  = (const float*)d_in[4];
    const float* W_exp = (const float*)d_in[5];
    const float* b_exp = (const float*)d_in[6];
    const float* Wg_sh = (const float*)d_in[7];
    const float* bg_sh = (const float*)d_in[8];
    const float* Wg_t1 = (const float*)d_in[9];
    const float* bg_t1 = (const float*)d_in[10];
    const float* Wg_t2 = (const float*)d_in[11];
    const float* bg_t2 = (const float*)d_in[12];
    const float* W_span = (const float*)d_in[13];
    const float* b_span = (const float*)d_in[14];
    const float* W_as   = (const float*)d_in[15];
    const float* b_as   = (const float*)d_in[16];
    const float* W_ae   = (const float*)d_in[17];
    const float* b_ae   = (const float*)d_in[18];
    float* out = (float*)d_out;

    char* ws = (char*)d_ws;
    float*          h   = (float*)(ws);                      // 3*4096*768*4  = 37,748,736
    float*          xf  = (float*)(ws + 37748736);           // 37,748,736
    unsigned short* xbv = (unsigned short*)(ws + 75497472);  // 18,874,368
    float*          e   = (float*)(ws + 94371840);           // 301,989,888
    float*          g   = (float*)(ws + 396361728);          // 917,504
    unsigned int*   wsw = (unsigned int*)(ws + 397279232);   // 56,623,104

    const int n_row_elems = ROWS * HH;                       // 3,145,728
    const int w_total = 6 * NT * KT * 32 * 8;                // 14,155,776
    const size_t w_layer_stride = (size_t)3 * NT * KT * 32 * 8;

    ple_convert_w<<<(w_total + 255) / 256, 256, 0, stream>>>(W_exp, wsw, w_total);
    ple_init_h<<<(n_row_elems + 255) / 256, 256, 0, stream>>>(share_hidden, span_hidden,
                                                              attr_hidden, h, n_row_elems);
    for (int l = 0; l < LL; ++l) {
        ple_layernorm<<<1536, 256, 0, stream>>>(h, ln_g + (size_t)l * 3 * HH,
                                                ln_b + (size_t)l * 3 * HH, xf, xbv);
        ple_gates<<<dim3(512, 3), 256, 0, stream>>>(
            xf,
            Wg_sh + (size_t)l * HH * 24, bg_sh + (size_t)l * 24,
            Wg_t1 + (size_t)l * HH * 16, bg_t1 + (size_t)l * 16,
            Wg_t2 + (size_t)l * HH * 16, bg_t2 + (size_t)l * 16, g);
        ple_gemm_experts<<<dim3(384, 12), 256, 0, stream>>>(
            xbv, wsw + (size_t)l * w_layer_stride, b_exp + (size_t)l * 3 * NN, e);
        ple_combine<<<4096, 256, 0, stream>>>(e, g, h);
    }
    ple_heads<<<4096, 96, 0, stream>>>(h, W_span, b_span, W_as, b_as, W_ae, b_ae, out);
}
